// GActor_78417512890496
// MI455X (gfx1250) — compile-verified, matrix operand order checked by reading, not run on hardware
//
#include <hip/hip_runtime.h>
#include <hip/hip_bf16.h>

typedef __attribute__((ext_vector_type(16))) _Float16 v16h;
typedef __attribute__((ext_vector_type(8)))  float    v8f;

#define H 128
#define F_IN 64

static inline int cdiv(int a, int b) { return (a + b - 1) / b; }

// ---------------- degree / norm precompute ----------------

__global__ void k_init_deg(float* __restrict__ deg, int N) {
  int i = blockIdx.x * blockDim.x + threadIdx.x;
  if (i < N) deg[i] = 1.0f;  // self-loop
}

__global__ void k_deg_accum(float* __restrict__ deg, const int* __restrict__ dst, int E) {
  int e = blockIdx.x * blockDim.x + threadIdx.x;
  if (e < E) atomicAdd(deg + dst[e], 1.0f);
}

__global__ void k_dinv(float* __restrict__ degdinv, int N) {
  int i = blockIdx.x * blockDim.x + threadIdx.x;
  if (i < N) {
    float d = degdinv[i];
    degdinv[i] = (d > 0.0f) ? rsqrtf(d) : 0.0f;
  }
}

__global__ void k_norm(float* __restrict__ normv, const int* __restrict__ src,
                       const int* __restrict__ dst, const float* __restrict__ dinv, int E) {
  int e = blockIdx.x * blockDim.x + threadIdx.x;
  if (e < E) normv[e] = dinv[src[e]] * dinv[dst[e]];
}

// ---------------- weight -> WMMA B-fragment (f16) ----------------
// Layout per ISA 7.12.2 (16-bit operand, 16x16x32):
// frag[kblk][nblk][lane][e]; element e of lane l maps to
//   k = kblk*32 + (e>>3)*16 + ((l>=16)?8:0) + (e&7),  n = nblk*16 + (l&15)

__global__ void k_wfrag(const float* __restrict__ W, _Float16* __restrict__ frag, int K) {
  int t = blockIdx.x * blockDim.x + threadIdx.x;
  if (t >= K * H) return;
  int e    = t & 15;
  int lane = (t >> 4) & 31;
  int nblk = (t >> 9) & 7;
  int kblk = t >> 12;
  int k = kblk * 32 + ((e >> 3) << 4) + ((lane >> 4) << 3) + (e & 7);
  int n = nblk * 16 + (lane & 15);
  frag[t] = (_Float16)W[k * H + n];
}

// ---------------- GEMM: out[N,128] = preop(in)[N,K] @ Wfrag ----------------
// One block = 8 waves, each wave owns a 16-row strip x all 128 cols (8 tiles).
// PREOP fuses previous layer's epilogue: relu(in + bias_prev[k]).

template <int K, bool PREOP>
__global__ __launch_bounds__(256) void k_gemm(const float* __restrict__ in,
                                              const _Float16* __restrict__ wfrag,
                                              const float* __restrict__ bias,
                                              float* __restrict__ out, int N) {
  const int wave = threadIdx.x >> 5;
  const int lane = threadIdx.x & 31;
  const int m    = lane & 15;
  const int kb   = (lane >> 4) << 3;  // 0 or 8
  const int rowtile = blockIdx.x * 128 + wave * 16;

  int row  = rowtile + m;
  int lrow = row < N ? row : N - 1;  // clamp loads; stores guarded (EXEC stays full for WMMA)
  const float* arow = in + (size_t)lrow * K;

  v8f acc[8] = {};

#pragma unroll
  for (int kc = 0; kc < K; kc += 32) {
    v16h a;
    const float* p0 = arow + kc + kb;
#pragma unroll
    for (int e = 0; e < 8; ++e) {
      float x0 = p0[e];
      float x1 = p0[e + 16];
      if (PREOP) {
        x0 = fmaxf(x0 + bias[kc + kb + e], 0.0f);
        x1 = fmaxf(x1 + bias[kc + kb + e + 16], 0.0f);
      }
      a[e]     = (_Float16)x0;
      a[e + 8] = (_Float16)x1;
    }
    const _Float16* wk = wfrag + ((size_t)(kc >> 5) << 12) + ((size_t)lane << 4);
#pragma unroll
    for (int nb = 0; nb < 8; ++nb) {
      v16h b = *(const v16h*)(wk + ((size_t)nb << 9));
      acc[nb] = __builtin_amdgcn_wmma_f32_16x16x32_f16(
          false, a, false, b, (short)0, acc[nb], false, false);
    }
  }

  // C layout: VGPR r: lanes 0-15 -> M=r, lanes 16-31 -> M=8+r; N = lane&15
  const int col   = lane & 15;
  const int rhalf = (lane >> 4) << 3;
#pragma unroll
  for (int nb = 0; nb < 8; ++nb) {
#pragma unroll
    for (int r = 0; r < 8; ++r) {
      int orow = rowtile + rhalf + r;
      if (orow < N) out[(size_t)orow * H + nb * 16 + col] = acc[nb][r];
    }
  }
}

// ---------------- self-loop init: agg = dinv[i]^2 * hlin ----------------

__global__ void k_self_init(float* __restrict__ agg, const float* __restrict__ hlin,
                            const float* __restrict__ dinv, int N) {
  int t = blockIdx.x * blockDim.x + threadIdx.x;  // over N*32 float4s
  if (t >= N * 32) return;
  int i = t >> 5;
  float d = dinv[i];
  float s = d * d;
  float4 v = ((const float4*)hlin)[t];
  float4 o = {s * v.x, s * v.y, s * v.z, s * v.w};
  ((float4*)agg)[t] = o;
}

// ---------------- edge scatter-add (128-wide) ----------------
// thread = (edge, quarter); each handles 32 floats via 8 float4 loads.

__global__ void k_scatter(float* __restrict__ agg, const float* __restrict__ hlin,
                          const int* __restrict__ src, const int* __restrict__ dst,
                          const float* __restrict__ normv, int E) {
  int t = blockIdx.x * blockDim.x + threadIdx.x;
  if (t >= E * 4) return;
  int e = t >> 2, q = t & 3;
  int s = src[e], d = dst[e];
  float nm = normv[e];
  const float4* ps = (const float4*)(hlin + (size_t)s * H + q * 32);
  float* pd = agg + (size_t)d * H + q * 32;
#pragma unroll
  for (int i = 0; i < 8; ++i) {
    float4 v = ps[i];
    atomicAdd(pd + i * 4 + 0, nm * v.x);
    atomicAdd(pd + i * 4 + 1, nm * v.y);
    atomicAdd(pd + i * 4 + 2, nm * v.z);
    atomicAdd(pd + i * 4 + 3, nm * v.w);
  }
}

// ---------------- final layer: dot(relu(agg+b2[2]), W3), self-loop + b3 ----------------

__global__ void k_dot_final(const float* __restrict__ agg, const float* __restrict__ bprev,
                            const float* __restrict__ W3, const float* __restrict__ b3,
                            const float* __restrict__ dinv, float* __restrict__ hval,
                            float* __restrict__ out, int N) {
  int i = blockIdx.x * blockDim.x + threadIdx.x;
  if (i >= N) return;
  const float* row = agg + (size_t)i * H;
  float acc = 0.0f;
#pragma unroll 8
  for (int f = 0; f < H; ++f) acc += fmaxf(row[f] + bprev[f], 0.0f) * W3[f];
  hval[i] = acc;
  float d = dinv[i];
  out[i] = d * d * acc + b3[0];
}

__global__ void k_scatter1(float* __restrict__ out, const float* __restrict__ hval,
                           const int* __restrict__ src, const int* __restrict__ dst,
                           const float* __restrict__ normv, int E) {
  int e = blockIdx.x * blockDim.x + threadIdx.x;
  if (e < E) atomicAdd(out + dst[e], normv[e] * hval[src[e]]);
}

// ---------------- launch ----------------

extern "C" void kernel_launch(void* const* d_in, const int* in_sizes, int n_in,
                              void* d_out, int out_size, void* d_ws, size_t ws_size,
                              hipStream_t stream) {
  const float* x  = (const float*)d_in[0];
  const int*   ei = (const int*)d_in[1];
  const float* W1 = (const float*)d_in[2];
  const float* b1 = (const float*)d_in[3];
  const float* W2 = (const float*)d_in[4];
  const float* b2 = (const float*)d_in[5];
  const float* W3 = (const float*)d_in[6];
  const float* b3 = (const float*)d_in[7];
  float* outp = (float*)d_out;

  const int N = in_sizes[0] / F_IN;
  const int E = in_sizes[1] / 2;
  const int* src = ei;
  const int* dst = ei + E;

  // workspace carve (256B aligned)
  char* p = (char*)d_ws;
  auto alloc = [&](size_t bytes) -> void* {
    void* r = (void*)p;
    p += (bytes + 255) & ~(size_t)255;
    return r;
  };
  float*     dinv  = (float*)alloc((size_t)N * 4);          // deg then dinv in place
  float*     normv = (float*)alloc((size_t)E * 4);
  float*     hval  = (float*)alloc((size_t)N * 4);
  _Float16*  w1f   = (_Float16*)alloc((size_t)F_IN * H * 2);
  _Float16*  w2f   = (_Float16*)alloc((size_t)3 * H * H * 2);
  float*     Hlin  = (float*)alloc((size_t)N * H * 4);
  float*     agg   = (float*)alloc((size_t)N * H * 4);
  (void)ws_size; (void)n_in; (void)out_size;

  const int T = 256;

  // normalization
  k_init_deg<<<cdiv(N, T), T, 0, stream>>>(dinv, N);
  k_deg_accum<<<cdiv(E, T), T, 0, stream>>>(dinv, dst, E);
  k_dinv<<<cdiv(N, T), T, 0, stream>>>(dinv, N);
  k_norm<<<cdiv(E, T), T, 0, stream>>>(normv, src, dst, dinv, E);

  // weight fragment conversion (once per launch)
  k_wfrag<<<cdiv(F_IN * H, T), T, 0, stream>>>(W1, w1f, F_IN);
  for (int i = 0; i < 3; ++i)
    k_wfrag<<<cdiv(H * H, T), T, 0, stream>>>(W2 + (size_t)i * H * H,
                                              w2f + (size_t)i * H * H, H);

  // layer 1: Hlin = x @ W1 ; agg = D^-1/2 A D^-1/2 Hlin
  k_gemm<F_IN, false><<<cdiv(N, 128), 256, 0, stream>>>(x, w1f, nullptr, Hlin, N);
  k_self_init<<<cdiv(N * 32, T), T, 0, stream>>>(agg, Hlin, dinv, N);
  k_scatter<<<cdiv(E * 4, T), T, 0, stream>>>(agg, Hlin, src, dst, normv, E);

  // hidden layers: prologue fuses relu(agg + bias_prev)
  const float* bprev[3] = {b1, b2, b2 + H};
  for (int i = 0; i < 3; ++i) {
    k_gemm<H, true><<<cdiv(N, 128), 256, 0, stream>>>(
        agg, w2f + (size_t)i * H * H, bprev[i], Hlin, N);
    k_self_init<<<cdiv(N * 32, T), T, 0, stream>>>(agg, Hlin, dinv, N);
    k_scatter<<<cdiv(E * 4, T), T, 0, stream>>>(agg, Hlin, src, dst, normv, E);
  }

  // output layer: h = relu(agg + b2[2]); out = scatter(h @ W3) + b3
  k_dot_final<<<cdiv(N, T), T, 0, stream>>>(agg, b2 + 2 * H, W3, b3, dinv, hval, outp, N);
  k_scatter1<<<cdiv(E, T), T, 0, stream>>>(outp, hval, src, dst, normv, E);
}
